// MPNNPredictor_24283745091791
// MI455X (gfx1250) — matrix operand order repaired and checked by reading, not compiled, not run on hardware
//
#include <hip/hip_runtime.h>
#include <hip/hip_bf16.h>

#define N_NODES 32000
#define N_EDGES 80000
#define N_GRAPH 1000
#define D 64
#define EH 128

// swizzled B buffer: 256 We2 chunks + 2 be2 chunks, fragment-major
#define B_CHUNK_US (4 * 32 * 16)            // ushorts per K-chunk (4 N-tiles)
#define WE2S_US   (256 * B_CHUNK_US)        // 524288 ushorts = 1 MB
#define BSWZ_US   (258 * B_CHUNK_US)

typedef __attribute__((ext_vector_type(16))) __bf16 v16bf;
typedef __attribute__((ext_vector_type(8)))  unsigned u32x8;
typedef __attribute__((ext_vector_type(8)))  float v8f;

__device__ __forceinline__ unsigned short f2bf(float f) {
    unsigned u = __float_as_uint(f);
    u += 0x7FFFu + ((u >> 16) & 1u);          // RNE
    return (unsigned short)(u >> 16);
}
__device__ __forceinline__ unsigned pkbf(float lo, float hi) {
    return (unsigned)f2bf(lo) | ((unsigned)f2bf(hi) << 16);
}
// native CDNA5 packed bf16 multiply (VOP3P V_PK_MUL_BF16) via inline asm
__device__ __forceinline__ unsigned pkmul_bf16(unsigned a, unsigned b) {
    unsigned d;
    asm("v_pk_mul_bf16 %0, %1, %2" : "=v"(d) : "v"(a), "v"(b));
    return d;
}
__device__ __forceinline__ float sigmoidf(float x) { return 1.0f / (1.0f + expf(-x)); }

// A-fragment = packed-bf16 h half scaled by packed-duplicated re
__device__ __forceinline__ v16bf scale_frag(const u32x8& hpk, unsigned repk) {
    u32x8 t;
#pragma unroll
    for (int i = 0; i < 8; ++i) t[i] = pkmul_bf16(hpk[i], repk);
    return __builtin_bit_cast(v16bf, t);
}

// ---------------- prep kernels ----------------

__global__ void proj_kernel(const float* __restrict__ nf, const float* __restrict__ Wp,
                            const float* __restrict__ bp, float* __restrict__ h) {
    int idx = blockIdx.x * blockDim.x + threadIdx.x;
    if (idx >= N_NODES * D) return;
    int n = idx >> 6, d = idx & 63;
    float acc = bp[d];
    const float* row = nf + n * 74;
    for (int k = 0; k < 74; ++k) acc += row[k] * Wp[k * D + d];
    h[idx] = fmaxf(acc, 0.0f);
}

// r (bf16, duplicated into both halves of a u32) = relu(edge_feats @ We1 + be1)
__global__ void edge_hidden_kernel(const float* __restrict__ ef, const float* __restrict__ We1,
                                   const float* __restrict__ be1, unsigned* __restrict__ r_dup) {
    int idx = blockIdx.x * blockDim.x + threadIdx.x;
    if (idx >= N_EDGES * EH) return;
    int e = idx >> 7, k = idx & 127;
    float acc = be1[k];
    const float* row = ef + e * 12;
    for (int j = 0; j < 12; ++j) acc += row[j] * We1[j * EH + k];
    unsigned b = f2bf(fmaxf(acc, 0.0f));
    r_dup[idx] = b | (b << 16);
}

// We2 [128,4096] -> fragment-major bf16 at Bswz[0..): idx = (((kk*4+nt)*32+lane)*16+j)
__global__ void swz_We2_kernel(const float* __restrict__ We2, unsigned short* __restrict__ outp) {
    int idx = blockIdx.x * blockDim.x + threadIdx.x;
    if (idx >= WE2S_US) return;
    int j = idx & 15, lane = (idx >> 4) & 31, nt = (idx >> 9) & 3, kk = idx >> 11;
    int keh = kk >> 1, d0 = (kk & 1) << 5;
    int m = lane & 15, off = (lane >> 4) << 3;
    int koff = (j < 8) ? (off + j) : (16 + off + (j - 8));
    outp[idx] = f2bf(We2[keh * 4096 + (d0 + koff) * D + nt * 16 + m]);
}

// be2 viewed as [64,64] -> fragment-major at Bswz[WE2S_US..): 2 chunks
__global__ void swz_be2_kernel(const float* __restrict__ be2, unsigned short* __restrict__ outp) {
    int idx = blockIdx.x * blockDim.x + threadIdx.x;
    if (idx >= 2 * B_CHUNK_US) return;
    int j = idx & 15, lane = (idx >> 4) & 31, nt = (idx >> 9) & 3, hl = idx >> 11;
    int m = lane & 15, off = (lane >> 4) << 3;
    int koff = (j < 8) ? (off + j) : (16 + off + (j - 8));
    outp[WE2S_US + idx] = f2bf(be2[(hl * 32 + koff) * D + nt * 16 + m]);
}

// GRU weight [64,192] -> fragment-major: idx = (((kc*12+nt)*32+lane)*16+j)
__global__ void swz_w192_kernel(const float* __restrict__ W, unsigned short* __restrict__ outp) {
    int idx = blockIdx.x * blockDim.x + threadIdx.x;
    if (idx >= 2 * 12 * 32 * 16) return;
    int j = idx & 15, lane = (idx >> 4) & 31;
    int rest = idx >> 9;           // 0..23
    int nt = rest % 12, kc = rest / 12;
    int m = lane & 15, off = (lane >> 4) << 3;
    int koff = (j < 8) ? (off + j) : (16 + off + (j - 8));
    outp[idx] = f2bf(W[(kc * 32 + koff) * 192 + nt * 16 + m]);
}

__global__ void zero_kernel(float* __restrict__ p, int n) {
    int i = blockIdx.x * blockDim.x + threadIdx.x;
    if (i < n) p[i] = 0.0f;
}
__global__ void fill_bias_kernel(float* __restrict__ agg, const float* __restrict__ b_conv) {
    int idx = blockIdx.x * blockDim.x + threadIdx.x;
    if (idx >= N_NODES * D) return;
    agg[idx] = b_conv[idx & 63];
}
__global__ void copy_kernel(const float* __restrict__ in, float* __restrict__ out, int n) {
    int i = blockIdx.x * blockDim.x + threadIdx.x;
    if (i < n) out[i] = in[i];
}

// ---------------- fused edge-message WMMA GEMM ----------------
// msg[E,64] = (r_e (x) h_src[e]) @ We2_flat[8192,64] + h_src[e] @ be2_mat[64,64]
// Software-pipelined: B fragments of keh i+1 prefetched during WMMAs of keh i.
// The 129th chunk streamed is the be2 epilogue (re = 1.0).
#define MSG_WAVES 4
__global__ __launch_bounds__(MSG_WAVES * 32)
void msg_kernel(const float* __restrict__ h, const unsigned* __restrict__ r_dup,
                const unsigned short* __restrict__ Bswz,
                const int* __restrict__ src, float* __restrict__ msg) {
    const int wave = threadIdx.x >> 5;
    const int lane = threadIdx.x & 31;
    const int tileM = blockIdx.x * MSG_WAVES + wave;   // 0..4999
    const int e0 = tileM * 16;
    const int m   = lane & 15;
    const int off = (lane >> 4) << 3;

    // hoist this lane's 32 h values, pre-packed to bf16 pairs (loop-invariant over K)
    const float* hrow = h + src[e0 + m] * D;
    u32x8 ha[2];
#pragma unroll
    for (int g = 0; g < 8; ++g) {
        int base = (g >> 1) * 16 + off + (g & 1) * 4;
        float4 qv = *(const float4*)(hrow + base);
        ha[g >> 2][2 * (g & 3)]     = pkbf(qv.x, qv.y);
        ha[g >> 2][2 * (g & 3) + 1] = pkbf(qv.z, qv.w);
    }
    const unsigned* rrow = r_dup + (e0 + m) * EH;

    v8f acc[4];
    v8f zero8 = {0.f, 0.f, 0.f, 0.f, 0.f, 0.f, 0.f, 0.f};
#pragma unroll
    for (int nt = 0; nt < 4; ++nt) acc[nt] = zero8;

    // preload keh 0 fragments (f = hl*4 + nt)
    const unsigned short* Blane = Bswz + lane * 16;
    v16bf bc[8];
#pragma unroll
    for (int f = 0; f < 8; ++f)
        bc[f] = *(const v16bf*)(Blane + (f >> 2) * 2048 + (f & 3) * 512);
    const unsigned short* Bnext = Blane + 4096;    // keh 1

#pragma unroll 1
    for (int kb = 0; kb < 16; ++kb) {
        u32x8 rv = *(const u32x8*)(rrow + kb * 8);   // 8 pre-duplicated r values
#pragma unroll
        for (int ki = 0; ki < 8; ++ki) {
            // prefetch next keh's 8 B fragments (final one fetches the be2 chunks)
            v16bf bn[8];
#pragma unroll
            for (int f = 0; f < 8; ++f)
                bn[f] = *(const v16bf*)(Bnext + (f >> 2) * 2048 + (f & 3) * 512);
            const unsigned repk = rv[ki];
            v16bf a0 = scale_frag(ha[0], repk);      // 8x v_pk_mul_bf16
            v16bf a1 = scale_frag(ha[1], repk);
            acc[0] = __builtin_amdgcn_wmma_f32_16x16x32_bf16(false, a0, false, bc[0], (short)0, acc[0], false, false);
            acc[1] = __builtin_amdgcn_wmma_f32_16x16x32_bf16(false, a0, false, bc[1], (short)0, acc[1], false, false);
            acc[2] = __builtin_amdgcn_wmma_f32_16x16x32_bf16(false, a0, false, bc[2], (short)0, acc[2], false, false);
            acc[3] = __builtin_amdgcn_wmma_f32_16x16x32_bf16(false, a0, false, bc[3], (short)0, acc[3], false, false);
            acc[0] = __builtin_amdgcn_wmma_f32_16x16x32_bf16(false, a1, false, bc[4], (short)0, acc[0], false, false);
            acc[1] = __builtin_amdgcn_wmma_f32_16x16x32_bf16(false, a1, false, bc[5], (short)0, acc[1], false, false);
            acc[2] = __builtin_amdgcn_wmma_f32_16x16x32_bf16(false, a1, false, bc[6], (short)0, acc[2], false, false);
            acc[3] = __builtin_amdgcn_wmma_f32_16x16x32_bf16(false, a1, false, bc[7], (short)0, acc[3], false, false);
#pragma unroll
            for (int f = 0; f < 8; ++f) bc[f] = bn[f];
            Bnext += 4096;
        }
    }
    // be2 epilogue (re = 1): bc holds the be2 fragments, A = pre-packed h bits
    {
        v16bf a0 = __builtin_bit_cast(v16bf, ha[0]);
        v16bf a1 = __builtin_bit_cast(v16bf, ha[1]);
        acc[0] = __builtin_amdgcn_wmma_f32_16x16x32_bf16(false, a0, false, bc[0], (short)0, acc[0], false, false);
        acc[1] = __builtin_amdgcn_wmma_f32_16x16x32_bf16(false, a0, false, bc[1], (short)0, acc[1], false, false);
        acc[2] = __builtin_amdgcn_wmma_f32_16x16x32_bf16(false, a0, false, bc[2], (short)0, acc[2], false, false);
        acc[3] = __builtin_amdgcn_wmma_f32_16x16x32_bf16(false, a0, false, bc[3], (short)0, acc[3], false, false);
        acc[0] = __builtin_amdgcn_wmma_f32_16x16x32_bf16(false, a1, false, bc[4], (short)0, acc[0], false, false);
        acc[1] = __builtin_amdgcn_wmma_f32_16x16x32_bf16(false, a1, false, bc[5], (short)0, acc[1], false, false);
        acc[2] = __builtin_amdgcn_wmma_f32_16x16x32_bf16(false, a1, false, bc[6], (short)0, acc[2], false, false);
        acc[3] = __builtin_amdgcn_wmma_f32_16x16x32_bf16(false, a1, false, bc[7], (short)0, acc[3], false, false);
    }

    const int rbase = (lane >> 4) << 3;
#pragma unroll
    for (int nt = 0; nt < 4; ++nt) {
        const int col = nt * 16 + m;
#pragma unroll
        for (int j = 0; j < 8; ++j)
            msg[(e0 + rbase + j) * D + col] = acc[nt][j];
    }
}

// ---------------- scatter-add aggregation ----------------
__global__ void scatter_kernel(const float* __restrict__ msg, const int* __restrict__ dst,
                               float* __restrict__ agg) {
    int idx = blockIdx.x * blockDim.x + threadIdx.x;
    if (idx >= N_EDGES * D) return;
    int e = idx >> 6, d = idx & 63;
    atomicAdd(&agg[dst[e] * D + d], msg[idx]);
}

// ---------------- GRU-gate WMMA GEMM: out[N,192] = act(A[N,64]) @ W[64,192] ----------------
#define GG_WAVES 4
__global__ __launch_bounds__(GG_WAVES * 32)
void gru_gemm_kernel(const float* __restrict__ A, const unsigned short* __restrict__ Wsw,
                     float* __restrict__ out, int do_relu) {
    const int wave = threadIdx.x >> 5;
    const int lane = threadIdx.x & 31;
    const int tileM = blockIdx.x * GG_WAVES + wave;   // 0..1999
    const int r0 = tileM * 16;
    const int m   = lane & 15;
    const int off = (lane >> 4) << 3;

    const float* arow = A + (r0 + m) * D;
    u32x8 ha[2];
#pragma unroll
    for (int g = 0; g < 8; ++g) {
        int base = (g >> 1) * 16 + off + (g & 1) * 4;
        float4 qv = *(const float4*)(arow + base);
        if (do_relu) {
            qv.x = fmaxf(qv.x, 0.f); qv.y = fmaxf(qv.y, 0.f);
            qv.z = fmaxf(qv.z, 0.f); qv.w = fmaxf(qv.w, 0.f);
        }
        ha[g >> 2][2 * (g & 3)]     = pkbf(qv.x, qv.y);
        ha[g >> 2][2 * (g & 3) + 1] = pkbf(qv.z, qv.w);
    }

    v8f acc[12];
    v8f zero8 = {0.f, 0.f, 0.f, 0.f, 0.f, 0.f, 0.f, 0.f};
#pragma unroll
    for (int nt = 0; nt < 12; ++nt) acc[nt] = zero8;

#pragma unroll
    for (int kc = 0; kc < 2; ++kc) {
        v16bf a = __builtin_bit_cast(v16bf, ha[kc]);
        const unsigned short* Bb = Wsw + ((kc * 12) * 32 + lane) * 16;
#pragma unroll
        for (int nt = 0; nt < 12; ++nt) {
            v16bf b = *(const v16bf*)(Bb + nt * 512);
            acc[nt] = __builtin_amdgcn_wmma_f32_16x16x32_bf16(
                false, a, false, b, (short)0, acc[nt], false, false);
        }
    }

    const int rbase = (lane >> 4) << 3;
#pragma unroll
    for (int nt = 0; nt < 12; ++nt) {
        const int col = nt * 16 + m;
#pragma unroll
        for (int j = 0; j < 8; ++j)
            out[(r0 + rbase + j) * 192 + col] = acc[nt][j];
    }
}

// ---------------- GRU elementwise update (torch gate order r,z,n) ----------------
__global__ void gru_update_kernel(const float* __restrict__ gx, const float* __restrict__ gh,
                                  const float* __restrict__ bx, const float* __restrict__ bh,
                                  float* __restrict__ h) {
    int idx = blockIdx.x * blockDim.x + threadIdx.x;
    if (idx >= N_NODES * D) return;
    int n = idx >> 6, d = idx & 63;
    const float* gxr = gx + n * 192;
    const float* ghr = gh + n * 192;
    float r  = sigmoidf(gxr[d]       + bx[d]       + ghr[d]       + bh[d]);
    float z  = sigmoidf(gxr[64 + d]  + bx[64 + d]  + ghr[64 + d]  + bh[64 + d]);
    float nn = tanhf   (gxr[128 + d] + bx[128 + d] + r * (ghr[128 + d] + bh[128 + d]));
    h[idx] = (1.0f - z) * nn + z * h[idx];
}

// ---------------- Set2Set LSTM layer (torch gate order i,f,g,o) ----------------
__global__ void lstm_kernel(const float* __restrict__ xq, int FI,
                            const float* __restrict__ h_in, float* __restrict__ h_out,
                            float* __restrict__ c,
                            const float* __restrict__ Wx, const float* __restrict__ Wh,
                            const float* __restrict__ bx, const float* __restrict__ bh) {
    int idx = blockIdx.x * blockDim.x + threadIdx.x;
    if (idx >= N_GRAPH * D) return;
    int g = idx >> 6, j = idx & 63;
    float a0 = bx[j] + bh[j];
    float a1 = bx[64 + j] + bh[64 + j];
    float a2 = bx[128 + j] + bh[128 + j];
    float a3 = bx[192 + j] + bh[192 + j];
    const float* xr = xq + g * FI;
    for (int k = 0; k < FI; ++k) {
        float x = xr[k];
        const float* w = Wx + k * 256;
        a0 += x * w[j]; a1 += x * w[64 + j]; a2 += x * w[128 + j]; a3 += x * w[192 + j];
    }
    const float* hr = h_in + g * D;
    for (int k = 0; k < D; ++k) {
        float hv = hr[k];
        const float* w = Wh + k * 256;
        a0 += hv * w[j]; a1 += hv * w[64 + j]; a2 += hv * w[128 + j]; a3 += hv * w[192 + j];
    }
    float ci = sigmoidf(a0), cf = sigmoidf(a1), cg = tanhf(a2), co = sigmoidf(a3);
    float cn = cf * c[idx] + ci * cg;
    c[idx] = cn;
    h_out[idx] = co * tanhf(cn);
}

// ---------------- attention + readout: one wave (32 lanes) per graph ----------------
__global__ void attn_kernel(const float* __restrict__ h, const float* __restrict__ q,
                            float* __restrict__ q_star) {
    int gid  = blockIdx.x * blockDim.x + threadIdx.x;
    int g    = gid >> 5;
    int lane = gid & 31;
    if (g >= N_GRAPH) return;
    const float* hn = h + (g * 32 + lane) * D;   // node = g*32 + lane
    const float* qg = q + g * D;
    float e = 0.0f;
    for (int k = 0; k < D; ++k) e += hn[k] * qg[k];
    float mx = e;
    for (int s = 16; s; s >>= 1) mx = fmaxf(mx, __shfl_xor(mx, s, 32));
    float ex = expf(e - mx);
    float ss = ex;
    for (int s = 16; s; s >>= 1) ss += __shfl_xor(ss, s, 32);
    float alpha = ex / ss;
    q_star[g * 128 + lane]      = qg[lane];
    q_star[g * 128 + 32 + lane] = qg[32 + lane];
    for (int d = 0; d < D; ++d) {
        float v = alpha * hn[d];
        for (int s = 16; s; s >>= 1) v += __shfl_xor(v, s, 32);
        if (lane == 0) q_star[g * 128 + 64 + d] = v;
    }
}

// ---------------- prediction head ----------------
__global__ void head_kernel(const float* __restrict__ q_star,
                            const float* __restrict__ Wout1, const float* __restrict__ bout1,
                            const float* __restrict__ Wout2, const float* __restrict__ bout2,
                            float* __restrict__ pred) {
    int g = blockIdx.x * blockDim.x + threadIdx.x;
    if (g >= N_GRAPH) return;
    const float* qs = q_star + g * 128;
    float acc = bout2[0];
    for (int j = 0; j < D; ++j) {
        float t = bout1[j];
        for (int k = 0; k < 128; ++k) t += qs[k] * Wout1[k * D + j];
        acc += fmaxf(t, 0.0f) * Wout2[j];
    }
    pred[g] = acc;
}

// ---------------- host orchestration ----------------
extern "C" void kernel_launch(void* const* d_in, const int* in_sizes, int n_in,
                              void* d_out, int out_size, void* d_ws, size_t ws_size,
                              hipStream_t stream) {
    const float* node_feats = (const float*)d_in[0];
    const float* edge_feats = (const float*)d_in[1];
    const int*   src        = (const int*)d_in[2];
    const int*   dst        = (const int*)d_in[3];
    const float* Wp   = (const float*)d_in[5];
    const float* bp   = (const float*)d_in[6];
    const float* We1  = (const float*)d_in[7];
    const float* be1  = (const float*)d_in[8];
    const float* We2  = (const float*)d_in[9];
    const float* be2  = (const float*)d_in[10];
    const float* b_conv = (const float*)d_in[11];
    const float* gru_Wx = (const float*)d_in[12];
    const float* gru_Wh = (const float*)d_in[13];
    const float* gru_bx = (const float*)d_in[14];
    const float* gru_bh = (const float*)d_in[15];
    const float* Wout1  = (const float*)d_in[16];
    const float* bout1  = (const float*)d_in[17];
    const float* Wout2  = (const float*)d_in[18];
    const float* bout2  = (const float*)d_in[19];
    const float* lstm_Wx[3] = {(const float*)d_in[20], (const float*)d_in[24], (const float*)d_in[28]};
    const float* lstm_Wh[3] = {(const float*)d_in[21], (const float*)d_in[25], (const float*)d_in[29]};
    const float* lstm_bx[3] = {(const float*)d_in[22], (const float*)d_in[26], (const float*)d_in[30]};
    const float* lstm_bh[3] = {(const float*)d_in[23], (const float*)d_in[27], (const float*)d_in[31]};

    char* ws = (char*)d_ws;
    size_t off = 0;
    auto carve = [&](size_t bytes) { void* p = ws + off; off += (bytes + 255) & ~(size_t)255; return p; };
    float* h    = (float*)carve((size_t)N_NODES * D * 4);
    float* agg  = (float*)carve((size_t)N_NODES * D * 4);
    float* gx   = (float*)carve((size_t)N_NODES * 192 * 4);
    float* gh   = (float*)carve((size_t)N_NODES * 192 * 4);
    float* msg  = (float*)carve((size_t)N_EDGES * D * 4);
    unsigned* r_dup = (unsigned*)carve((size_t)N_EDGES * EH * 4);
    unsigned short* Bswz = (unsigned short*)carve((size_t)BSWZ_US * 2);  // We2 + be2 fragments
    unsigned short* WxS  = (unsigned short*)carve((size_t)2 * 12 * 32 * 16 * 2);
    unsigned short* WhS  = (unsigned short*)carve((size_t)2 * 12 * 32 * 16 * 2);
    float* hsA   = (float*)carve((size_t)3 * N_GRAPH * D * 4);
    float* hsB   = (float*)carve((size_t)3 * N_GRAPH * D * 4);
    float* cs    = (float*)carve((size_t)3 * N_GRAPH * D * 4);
    float* qstar = (float*)carve((size_t)N_GRAPH * 128 * 4);

    // ---- prep ----
    proj_kernel<<<(N_NODES * D + 255) / 256, 256, 0, stream>>>(node_feats, Wp, bp, h);
    edge_hidden_kernel<<<(N_EDGES * EH + 255) / 256, 256, 0, stream>>>(edge_feats, We1, be1, r_dup);
    swz_We2_kernel<<<(WE2S_US + 255) / 256, 256, 0, stream>>>(We2, Bswz);
    swz_be2_kernel<<<(2 * B_CHUNK_US + 255) / 256, 256, 0, stream>>>(be2, Bswz);
    swz_w192_kernel<<<(2 * 12 * 32 * 16 + 255) / 256, 256, 0, stream>>>(gru_Wx, WxS);
    swz_w192_kernel<<<(2 * 12 * 32 * 16 + 255) / 256, 256, 0, stream>>>(gru_Wh, WhS);

    // ---- 6 message-passing + GRU steps ----
    for (int s = 0; s < 6; ++s) {
        fill_bias_kernel<<<(N_NODES * D + 255) / 256, 256, 0, stream>>>(agg, b_conv);
        msg_kernel<<<(N_EDGES / 16) / MSG_WAVES, MSG_WAVES * 32, 0, stream>>>(
            h, r_dup, Bswz, src, msg);
        scatter_kernel<<<(N_EDGES * D + 255) / 256, 256, 0, stream>>>(msg, dst, agg);
        gru_gemm_kernel<<<(N_NODES / 16) / GG_WAVES, GG_WAVES * 32, 0, stream>>>(agg, WxS, gx, 1);
        gru_gemm_kernel<<<(N_NODES / 16) / GG_WAVES, GG_WAVES * 32, 0, stream>>>(h, WhS, gh, 0);
        gru_update_kernel<<<(N_NODES * D + 255) / 256, 256, 0, stream>>>(gx, gh, gru_bx, gru_bh, h);
    }

    // ---- Set2Set ----
    zero_kernel<<<(3 * N_GRAPH * D + 255) / 256, 256, 0, stream>>>(hsA, 3 * N_GRAPH * D);
    zero_kernel<<<(3 * N_GRAPH * D + 255) / 256, 256, 0, stream>>>(cs, 3 * N_GRAPH * D);
    zero_kernel<<<(N_GRAPH * 128 + 255) / 256, 256, 0, stream>>>(qstar, N_GRAPH * 128);
    for (int t = 0; t < 6; ++t) {
        float* hin  = (t & 1) ? hsB : hsA;
        float* hout = (t & 1) ? hsA : hsB;
        const float* xq = qstar;
        int FI = 128;
        for (int l = 0; l < 3; ++l) {
            lstm_kernel<<<(N_GRAPH * D + 255) / 256, 256, 0, stream>>>(
                xq, FI, hin + l * N_GRAPH * D, hout + l * N_GRAPH * D, cs + l * N_GRAPH * D,
                lstm_Wx[l], lstm_Wh[l], lstm_bx[l], lstm_bh[l]);
            xq = hout + l * N_GRAPH * D;
            FI = D;
        }
        attn_kernel<<<(N_GRAPH * 32 + 255) / 256, 256, 0, stream>>>(h, xq, qstar);
    }

    // ---- head: d_out = [pred(1000), q_star(128000)] ----
    float* out = (float*)d_out;
    head_kernel<<<(N_GRAPH + 255) / 256, 256, 0, stream>>>(qstar, Wout1, bout1, Wout2, bout2, out);
    copy_kernel<<<(N_GRAPH * 128 + 255) / 256, 256, 0, stream>>>(qstar, out + N_GRAPH, N_GRAPH * 128);
}
